// _MMLNorm_58420145160265
// MI455X (gfx1250) — matrix operand order repaired and checked by reading, not compiled, not run on hardware
//
#include <hip/hip_runtime.h>
#include <math.h>

typedef __attribute__((ext_vector_type(16))) _Float16 v16h;
typedef __attribute__((ext_vector_type(8)))  float    v8f;
typedef __attribute__((ext_vector_type(4)))  float    v4f;   // native vector (nontemporal-OK)

#define B_DIM 8
#define T_DIM 4096
#define D_DIM 1024
#define K_DIM 256

// ---------------------------------------------------------------------------
// Kernel 1: tar_bias[8,1024] = tanh(convInfo[8,256] @ lin_w[1024,256]^T + lin_b)
// One wave per 16-wide N tile (M padded 8->16 via lane mask). fp32 inputs are
// split into hi/lo f16 pairs; acc += Ahi*Bhi + Ahi*Blo + Alo*Bhi in f32 WMMA,
// giving ~fp32-exact results from the f16 matrix pipe.
// ---------------------------------------------------------------------------
__global__ __launch_bounds__(32) void condbias_wmma_kernel(
    const float* __restrict__ conv,    // [8, 256]
    const float* __restrict__ lin_w,   // [1024, 256]  (row n holds K values)
    const float* __restrict__ lin_b,   // [1024]
    float* __restrict__ tb)            // [8, 1024] output
{
    const int lane = threadIdx.x;      // 0..31, EXEC all-ones (no divergence)
    const int hs   = lane >> 4;        // half-wave select
    const int l    = lane & 15;
    const int n    = blockIdx.x * 16 + l;

    // M-padding: rows 8..15 of A are zero (masked multiply, no branches -> no
    // EXEC divergence before the WMMA, and no OOB reads).
    const float amask = (l < B_DIM) ? 1.0f : 0.0f;
    const float* __restrict__ arow_p = conv  + (size_t)(l & (B_DIM - 1)) * K_DIM;
    const float* __restrict__ bcol_p = lin_w + (size_t)n * K_DIM;

    v8f acc = {};

    #pragma unroll
    for (int kt = 0; kt < K_DIM; kt += 32) {
        v16h a_hi, a_lo, b_hi, b_lo;

        // A fragment, 16x32, this code's contiguous k order (same as B):
        //   lanes 0-15 : M=l, halves 0..15 = K+0..15
        //   lanes 16-31: M=l, halves 0..15 = K+16..31
        const int ka = kt + hs * 16;
        #pragma unroll
        for (int h = 0; h < 8; ++h) {
            float x0 = arow_p[ka + h]      * amask;
            float x1 = arow_p[ka + 8 + h] * amask;
            _Float16 h0 = (_Float16)x0, h1 = (_Float16)x1;
            a_hi[h]     = h0;  a_lo[h]     = (_Float16)(x0 - (float)h0);
            a_hi[8 + h] = h1;  a_lo[8 + h] = (_Float16)(x1 - (float)h1);
        }

        // B fragment, 32x16, this code's contiguous k order (same as A):
        //   lanes 0-15 : N=l, halves 0..15 = K+0..15
        //   lanes 16-31: N=l, halves 0..15 = K+16..31
        const int kb = kt + hs * 16;
        #pragma unroll
        for (int h = 0; h < 16; ++h) {
            float x = bcol_p[kb + h];
            _Float16 hh = (_Float16)x;
            b_hi[h] = hh;
            b_lo[h] = (_Float16)(x - (float)hh);
        }

        acc = __builtin_amdgcn_wmma_f32_16x16x32_f16(false, a_hi, false, b_hi,
                                                     (short)0, acc, false, false);
        acc = __builtin_amdgcn_wmma_f32_16x16x32_f16(false, a_hi, false, b_lo,
                                                     (short)0, acc, false, false);
        acc = __builtin_amdgcn_wmma_f32_16x16x32_f16(false, a_lo, false, b_hi,
                                                     (short)0, acc, false, false);
    }

    // C/D 16x16 f32 layout: VGPR v, lane -> M = v + 8*hs, N = lane&15.
    const float lb = lin_b[n];
    #pragma unroll
    for (int v = 0; v < 8; ++v) {
        int m = v + hs * 8;
        if (m < B_DIM) tb[(size_t)m * D_DIM + n] = tanhf(acc[v] + lb);
    }
}

// ---------------------------------------------------------------------------
// Kernel 2: fused LayerNorm + conditional-bias + second affine, one streaming
// pass. One wave32 per row of 1024; each lane owns 32 elements (8 x v4f).
// Non-temporal on the streamed 256 MiB; regular (cached) loads for the
// L2-resident weight/bias/tar_bias tables. Bandwidth-bound: ~11 us floor.
// ---------------------------------------------------------------------------
__global__ __launch_bounds__(256) void ln_condbias_kernel(
    const float* __restrict__ x,       // [8*4096, 1024]
    const float* __restrict__ w,       // [1024]
    const float* __restrict__ b,       // [1024]
    const float* __restrict__ tb,      // [8, 1024]
    float* __restrict__ out)           // [8*4096, 1024]
{
    const int row   = blockIdx.x * blockDim.y + threadIdx.y;   // 0..32767
    const int batch = row >> 12;                               // row / 4096
    const int lane  = threadIdx.x;                             // 0..31

    const v4f* __restrict__ xr  = (const v4f*)(x  + (size_t)row * D_DIM);
    v4f*       __restrict__ orr = (v4f*)      (out + (size_t)row * D_DIM);
    const v4f* __restrict__ wv4 = (const v4f*)w;
    const v4f* __restrict__ bv4 = (const v4f*)b;
    const v4f* __restrict__ tv4 = (const v4f*)(tb + (size_t)batch * D_DIM);

    v4f v[8];
    float s = 0.0f, s2 = 0.0f;
    #pragma unroll
    for (int i = 0; i < 8; ++i) {
        v[i] = __builtin_nontemporal_load(&xr[lane + 32 * i]);
        s  += v[i].x + v[i].y + v[i].z + v[i].w;
        s2 += v[i].x * v[i].x + v[i].y * v[i].y
            + v[i].z * v[i].z + v[i].w * v[i].w;
    }

    // wave32 butterfly reduction
    #pragma unroll
    for (int off = 16; off >= 1; off >>= 1) {
        s  += __shfl_xor(s,  off, 32);
        s2 += __shfl_xor(s2, off, 32);
    }

    const float mean = s * (1.0f / D_DIM);
    const float var  = s2 * (1.0f / D_DIM) - mean * mean;
    const float rstd = rsqrtf(var + 1e-5f);

    #pragma unroll
    for (int i = 0; i < 8; ++i) {
        const int c4 = lane + 32 * i;
        const v4f wv = wv4[c4];
        const v4f bv = bv4[c4];
        const v4f tv = tv4[c4];
        v4f o;
        o.x = ((v[i].x - mean) * rstd * wv.x + bv.x + tv.x) * wv.x + bv.x;
        o.y = ((v[i].y - mean) * rstd * wv.y + bv.y + tv.y) * wv.y + bv.y;
        o.z = ((v[i].z - mean) * rstd * wv.z + bv.z + tv.z) * wv.z + bv.z;
        o.w = ((v[i].w - mean) * rstd * wv.w + bv.w + tv.w) * wv.w + bv.w;
        __builtin_nontemporal_store(o, &orr[c4]);
    }
}

// ---------------------------------------------------------------------------
extern "C" void kernel_launch(void* const* d_in, const int* in_sizes, int n_in,
                              void* d_out, int out_size, void* d_ws, size_t ws_size,
                              hipStream_t stream) {
    const float* input  = (const float*)d_in[0];  // [8,4096,1024]
    const float* conv   = (const float*)d_in[1];  // [8,256]
    const float* weight = (const float*)d_in[2];  // [1024]
    const float* bias   = (const float*)d_in[3];  // [1024]
    const float* lin_w  = (const float*)d_in[4];  // [1024,256]
    const float* lin_b  = (const float*)d_in[5];  // [1024]
    float* out = (float*)d_out;
    float* tb  = (float*)d_ws;                    // 8*1024 floats = 32 KB scratch

    // Phase 1: tiny WMMA GEMM + tanh -> tar_bias (64 waves total)
    condbias_wmma_kernel<<<D_DIM / 16, 32, 0, stream>>>(conv, lin_w, lin_b, tb);

    // Phase 2: streaming fused LayerNorm (stream-ordered after phase 1)
    dim3 blk(32, 8);
    const int rows = B_DIM * T_DIM;               // 32768
    ln_condbias_kernel<<<rows / 8, blk, 0, stream>>>(input, weight, bias, tb, out);
}
